// ParallelSimplifiedHybridLayer_14310831030645
// MI455X (gfx1250) — compile-verified, matrix operand order checked by reading, not run on hardware
//
#include <hip/hip_runtime.h>
#include <hip/hip_bf16.h>
#include <math.h>

// ---------------------------------------------------------------------------
// MI455X (gfx1250) implementation of the hybrid transformer block.
// All matmuls (QKV, Q@K^T, P@V, Wo, Wsp, Wt1, Wt2, Wf1, Wf2) run on the
// CDNA5 matrix pipes via v_wmma_f32_16x16x32_bf16 (bf16 in, f32 accumulate).
// GEMM B-tiles are staged into LDS with the gfx1250 async global->LDS copy
// path (ASYNCcnt), double-buffered to overlap copy with WMMA. Attention is
// flash-style (online softmax, causal). The decay einsum is a linear
// recurrence scan.
// ---------------------------------------------------------------------------

typedef __attribute__((ext_vector_type(16))) __bf16 v16bf;
typedef __attribute__((ext_vector_type(8)))  __bf16 bf16x8;
typedef __attribute__((ext_vector_type(8)))  float  v8f;
typedef __attribute__((ext_vector_type(4)))  int    v4i;

union Frag { v16bf v; bf16x8 h[2]; };

#define WMMA_BF16(a, b, c) \
    __builtin_amdgcn_wmma_f32_16x16x32_bf16(false, (a), false, (b), (short)0, (c), false, false)

#define AS1 __attribute__((address_space(1)))
#define AS3 __attribute__((address_space(3)))

// 16-byte async copy global -> LDS (ASYNCcnt path); falls back to a regular
// load+ds_store if this toolchain lacks the builtin (still LDS-staged).
// Builtin signature (from clang diagnostic): (AS1 int4*, AS3 int4*, imm, cpol)
__device__ __forceinline__ void async_copy16(const __bf16* g, __bf16* l) {
#if __has_builtin(__builtin_amdgcn_global_load_async_to_lds_b128)
    __builtin_amdgcn_global_load_async_to_lds_b128((AS1 v4i*)g, (AS3 v4i*)l, 0, 0);
#else
    *(bf16x8*)l = *(const bf16x8*)g;
#endif
}

__device__ __forceinline__ void wait_async0() {
#if __has_builtin(__builtin_amdgcn_s_wait_asynccnt)
    __builtin_amdgcn_s_wait_asynccnt(0);
#else
    __builtin_amdgcn_s_waitcnt(0);  // conservative fallback
#endif
}

__device__ __forceinline__ float gelu_f(float x) {
    // exact GELU (approximate=False): 0.5*x*(1+erf(x/sqrt(2)))
    return 0.5f * x * (1.0f + erff(x * 0.70710678118654752f));
}

// ---------------------------------------------------------------------------
// Weight pack: W [K,N] f32 (row-major, as in  x @ W)  ->  Wt [N,K] bf16.
// With [N,K] layout, B-matrix fragments load with the same contiguous pattern
// as A-matrix fragments (ISA 7.12.2, 16-bit A 16x32 layout).
// ---------------------------------------------------------------------------
__global__ void pack_wt(const float* __restrict__ W, __bf16* __restrict__ Wt,
                        int K, int N) {
    size_t total = (size_t)K * N;
    for (size_t i = (size_t)blockIdx.x * blockDim.x + threadIdx.x; i < total;
         i += (size_t)gridDim.x * blockDim.x) {
        int k = (int)(i % K);
        int n = (int)(i / K);
        Wt[i] = (__bf16)W[(size_t)k * N + n];
    }
}

__global__ void f32_to_bf16(const float* __restrict__ in, __bf16* __restrict__ out,
                            size_t n) {
    for (size_t i = (size_t)blockIdx.x * blockDim.x + threadIdx.x; i < n;
         i += (size_t)gridDim.x * blockDim.x) {
        out[i] = (__bf16)in[i];
    }
}

// ---------------------------------------------------------------------------
// Tiled WMMA GEMM:  C[M,N] = act(A[M,K] @ Bt[N,K]^T + bias[N])
// Block = 256 threads = 8 waves; each wave owns a 16(M) x 64(N) tile, so the
// 64x32 B-tile per k-step is shared by all 8 waves: it is staged into LDS
// via async global->LDS b128 copies (one per thread), double-buffered so the
// next tile's copy overlaps this tile's WMMAs. grid = (N/64, M/128).
// act: 0 = none, 1 = exact GELU.
// ---------------------------------------------------------------------------
__global__ __launch_bounds__(256) void gemm_bf16(
    const __bf16* __restrict__ A, const __bf16* __restrict__ Bt,
    const float* __restrict__ bias, float* __restrict__ C,
    int M, int N, int K, int act) {
    __shared__ __bf16 bsm[2][64][32];      // double-buffered 64x32 B tile (8KB)

    const int lane = threadIdx.x & 31;
    const int wv   = threadIdx.x >> 5;
    const int l15  = lane & 15;
    const int kh   = lane >> 4;

    const int mt = blockIdx.y * 128 + wv * 16;   // 16-row M tile of this wave
    const int nb = blockIdx.x * 64;              // 64-col N tile of this block

    v8f acc[4] = {v8f{}, v8f{}, v8f{}, v8f{}};

    const __bf16* arow = A + (size_t)(mt + l15) * K;

    // cooperative B-tile copy: 256 threads x 16B = 4KB = 64 rows x 32 bf16
    const int crow  = threadIdx.x >> 2;          // 0..63: B row within tile
    const int cpart = (threadIdx.x & 3) * 8;     // 8-elem chunk within row
    const __bf16* gB = Bt + (size_t)(nb + crow) * K + cpart;

    // prologue: stage k-step 0 into buffer 0
    async_copy16(gB, &bsm[0][crow][cpart]);
    wait_async0();
    __syncthreads();

    const int nsteps = K >> 5;
    for (int i = 0; i < nsteps; ++i) {
        const int kk  = i << 5;
        const int cur = i & 1;
        const int nxt = cur ^ 1;
        // kick off next tile's async copy (overlaps the WMMAs below)
        if (i + 1 < nsteps)
            async_copy16(gB + (size_t)(kk + 32), &bsm[nxt][crow][cpart]);

        // A fragment (16x32 bf16) straight from global (per-wave unique):
        // elems 0..7 -> K = kk + 8*kh .. ; elems 8..15 -> K = kk+16+8*kh ..
        Frag af;
        af.h[0] = *(const bf16x8*)(arow + kk + 8 * kh);
        af.h[1] = *(const bf16x8*)(arow + kk + 16 + 8 * kh);
#pragma unroll
        for (int t = 0; t < 4; ++t) {
            const __bf16* brow = &bsm[cur][t * 16 + l15][0];
            Frag bf_;
            bf_.h[0] = *(const bf16x8*)(brow + 8 * kh);
            bf_.h[1] = *(const bf16x8*)(brow + 16 + 8 * kh);
            acc[t] = WMMA_BF16(af.v, bf_.v, acc[t]);
        }
        wait_async0();     // next tile landed
        __syncthreads();   // everyone done reading cur; nxt visible to all
    }

    // C/D layout: vgpr r, lane -> row = r + 8*kh, col = l15 (ISA 7.12.2)
#pragma unroll
    for (int t = 0; t < 4; ++t) {
        const int n = nb + t * 16 + l15;
        const float bv = bias[n];
#pragma unroll
        for (int r = 0; r < 8; ++r) {
            const int m = mt + r + 8 * kh;
            float v = acc[t][r] + bv;
            if (act == 1) v = gelu_f(v);
            C[(size_t)m * N + n] = v;
        }
    }
}

// ---------------------------------------------------------------------------
// Split qkv [B,T,3D] f32 into head-major bf16 buffers:
//   Qb [B*H, T, 64]  (pre-scaled by 1/sqrt(64) = 0.125)
//   Kb [B*H, T, 64]
//   Vt [B*H, 64, T]  (transposed so P@V B-fragments load contiguously)
// ---------------------------------------------------------------------------
__global__ void pack_qkv(const float* __restrict__ qkv,
                         __bf16* __restrict__ Qb, __bf16* __restrict__ Kb,
                         __bf16* __restrict__ Vt) {
    const int T = 2048, D = 512;
    size_t total = (size_t)4 * T * D;
    for (size_t i = (size_t)blockIdx.x * blockDim.x + threadIdx.x; i < total;
         i += (size_t)gridDim.x * blockDim.x) {
        int d = (int)(i % D);
        size_t bt = i / D;
        int t = (int)(bt % T);
        int b = (int)(bt / T);
        int h = d >> 6;
        int dd = d & 63;
        const float* row = qkv + bt * (3 * D);
        size_t qi = (((size_t)(b * 8 + h)) * T + t) * 64 + dd;
        Qb[qi] = (__bf16)(row[d] * 0.125f);
        Kb[qi] = (__bf16)(row[D + d]);
        Vt[((size_t)(b * 8 + h) * 64 + dd) * T + t] = (__bf16)(row[2 * D + d]);
    }
}

// ---------------------------------------------------------------------------
// Flash attention (causal), one wave per 16-query tile, hd = 64, T = 2048.
// S = (Q/8) @ K^T via 2 WMMAs per 16-key group; online softmax with
// half-wave shuffles (C-layout rows split lanes 0-15 / 16-31); P is staged
// through LDS to re-shape C-layout -> A-layout; O += P @ V via 4 WMMAs.
// grid = (T/128, B*H), block = 256 (8 waves).
// ---------------------------------------------------------------------------
__global__ __launch_bounds__(256) void flash_attn(
    const __bf16* __restrict__ Qb, const __bf16* __restrict__ Kb,
    const __bf16* __restrict__ Vt, float* __restrict__ ao) {
    const int T = 2048, HD = 64, D = 512;
    __shared__ __bf16 pbuf[8][16][32];   // per-wave 16x32 P tile

    const int lane = threadIdx.x & 31;
    const int wv   = threadIdx.x >> 5;
    const int l15  = lane & 15;
    const int kh   = lane >> 4;

    const int bh    = blockIdx.y;                  // b*H + h
    const int qbase = (blockIdx.x * 8 + wv) * 16;  // query tile start

    const __bf16* Qp = Qb + ((size_t)bh * T + qbase) * HD;
    const __bf16* Kp = Kb + (size_t)bh * T * HD;
    const __bf16* Vp = Vt + (size_t)bh * HD * T;

    // Q fragments (dims 0..31 and 32..63), row = qbase + l15
    Frag qf0, qf1;
    {
        const __bf16* qrow = Qp + (size_t)l15 * HD;
        qf0.h[0] = *(const bf16x8*)(qrow + 8 * kh);
        qf0.h[1] = *(const bf16x8*)(qrow + 16 + 8 * kh);
        qf1.h[0] = *(const bf16x8*)(qrow + 32 + 8 * kh);
        qf1.h[1] = *(const bf16x8*)(qrow + 48 + 8 * kh);
    }

    v8f oacc[4] = {v8f{}, v8f{}, v8f{}, v8f{}};
    float m8[8], l8[8];
#pragma unroll
    for (int r = 0; r < 8; ++r) { m8[r] = -1e30f; l8[r] = 0.0f; }

    const int kb_last = qbase & ~31;   // last 32-key block in causal region
    for (int kb = 0; kb <= kb_last; kb += 32) {
        // ---- scores for two 16-key groups ----
        v8f s[2];
#pragma unroll
        for (int gk = 0; gk < 2; ++gk) {
            const __bf16* krow = Kp + (size_t)(kb + gk * 16 + l15) * HD;
            Frag kf0, kf1;
            kf0.h[0] = *(const bf16x8*)(krow + 8 * kh);
            kf0.h[1] = *(const bf16x8*)(krow + 16 + 8 * kh);
            kf1.h[0] = *(const bf16x8*)(krow + 32 + 8 * kh);
            kf1.h[1] = *(const bf16x8*)(krow + 48 + 8 * kh);
            v8f c = {};
            c = WMMA_BF16(qf0.v, kf0.v, c);
            c = WMMA_BF16(qf1.v, kf1.v, c);
            s[gk] = c;
        }
        // ---- causal mask + online softmax, per C-layout row r ----
        const int key0 = kb + l15;
        const int key1 = kb + 16 + l15;
#pragma unroll
        for (int r = 0; r < 8; ++r) {
            const int q = qbase + r + 8 * kh;
            float s0 = (key0 <= q) ? s[0][r] : -1e30f;
            float s1 = (key1 <= q) ? s[1][r] : -1e30f;
            float mx = fmaxf(s0, s1);
#pragma unroll
            for (int o = 8; o > 0; o >>= 1) mx = fmaxf(mx, __shfl_xor(mx, o, 16));
            const float nm = fmaxf(m8[r], mx);
            const float p0 = __expf(s0 - nm);
            const float p1 = __expf(s1 - nm);
            float rs = p0 + p1;
#pragma unroll
            for (int o = 8; o > 0; o >>= 1) rs += __shfl_xor(rs, o, 16);
            const float alpha = __expf(m8[r] - nm);
            l8[r] = l8[r] * alpha + rs;
            m8[r] = nm;
#pragma unroll
            for (int t = 0; t < 4; ++t) oacc[t][r] *= alpha;
            // stash P in LDS (row = r + 8*kh, cols l15 / 16+l15)
            pbuf[wv][r + 8 * kh][l15]      = (__bf16)p0;
            pbuf[wv][r + 8 * kh][16 + l15] = (__bf16)p1;
        }
        // ---- reload P as an A-layout 16x32 fragment (wave-private LDS) ----
        Frag pf;
        {
            const __bf16* prow = &pbuf[wv][l15][0];
            pf.h[0] = *(const bf16x8*)(prow + 8 * kh);
            pf.h[1] = *(const bf16x8*)(prow + 16 + 8 * kh);
        }
        // ---- O += P @ V  (V^T rows = head dims, contiguous over keys) ----
#pragma unroll
        for (int t = 0; t < 4; ++t) {
            const __bf16* vrow = Vp + (size_t)(t * 16 + l15) * T + kb;
            Frag vf;
            vf.h[0] = *(const bf16x8*)(vrow + 8 * kh);
            vf.h[1] = *(const bf16x8*)(vrow + 16 + 8 * kh);
            oacc[t] = WMMA_BF16(pf.v, vf.v, oacc[t]);
        }
    }

    // ---- normalize and scatter into ao [B,T,D] (head-interleaved) ----
    const int b = bh >> 3;
    const int h = bh & 7;
#pragma unroll
    for (int r = 0; r < 8; ++r) {
        const float inv = 1.0f / l8[r];
        const int q = qbase + r + 8 * kh;
        float* orow = ao + ((size_t)b * T + q) * D + h * HD;
#pragma unroll
        for (int t = 0; t < 4; ++t) orow[t * 16 + l15] = oacc[t][r] * inv;
    }
}

// ---------------------------------------------------------------------------
// LayerNorm over D=512, optional residual: out = LN(x (+ res)) * g + b
// One block (128 threads, float4 per thread) per row.
// ---------------------------------------------------------------------------
__global__ __launch_bounds__(128) void ln_fused(
    const float* __restrict__ xin, const float* __restrict__ res,
    const float* __restrict__ g, const float* __restrict__ bta,
    float* __restrict__ out) {
    const int D = 512;
    const size_t row = blockIdx.x;
    const float4 vx = ((const float4*)(xin + row * D))[threadIdx.x];
    float4 v = vx;
    if (res != nullptr) {
        const float4 w = ((const float4*)(res + row * D))[threadIdx.x];
        v.x += w.x; v.y += w.y; v.z += w.z; v.w += w.w;
    }
    float s  = v.x + v.y + v.z + v.w;
    float s2 = v.x * v.x + v.y * v.y + v.z * v.z + v.w * v.w;
#pragma unroll
    for (int o = 16; o > 0; o >>= 1) {
        s  += __shfl_xor(s, o, 32);
        s2 += __shfl_xor(s2, o, 32);
    }
    __shared__ float sh[2][4];
    const int wv = threadIdx.x >> 5, lane = threadIdx.x & 31;
    if (lane == 0) { sh[0][wv] = s; sh[1][wv] = s2; }
    __syncthreads();
    s  = sh[0][0] + sh[0][1] + sh[0][2] + sh[0][3];
    s2 = sh[1][0] + sh[1][1] + sh[1][2] + sh[1][3];
    const float mu   = s * (1.0f / 512.0f);
    const float var  = s2 * (1.0f / 512.0f) - mu * mu;
    const float rstd = rsqrtf(var + 1e-5f);
    const float4 gg = ((const float4*)g)[threadIdx.x];
    const float4 bb = ((const float4*)bta)[threadIdx.x];
    float4 o4;
    o4.x = (v.x - mu) * rstd * gg.x + bb.x;
    o4.y = (v.y - mu) * rstd * gg.y + bb.y;
    o4.z = (v.z - mu) * rstd * gg.z + bb.z;
    o4.w = (v.w - mu) * rstd * gg.w + bb.w;
    ((float4*)(out + row * D))[threadIdx.x] = o4;
}

// ---------------------------------------------------------------------------
// Causal depthwise conv1d, kernel 3, left pad 2:
//   out[b,t,d] = w[d,0]*in[t-2] + w[d,1]*in[t-1] + w[d,2]*in[t] + cb[d]
// ---------------------------------------------------------------------------
__global__ void dwconv3(const float* __restrict__ in, const float* __restrict__ w,
                        const float* __restrict__ cb, float* __restrict__ out) {
    const int T = 2048, D = 512;
    size_t total = (size_t)4 * T * D;
    for (size_t i = (size_t)blockIdx.x * blockDim.x + threadIdx.x; i < total;
         i += (size_t)gridDim.x * blockDim.x) {
        int d = (int)(i % D);
        size_t bt = i / D;
        int t = (int)(bt % T);
        float a0 = (t >= 2) ? in[i - 2 * (size_t)D] : 0.0f;
        float a1 = (t >= 1) ? in[i - (size_t)D] : 0.0f;
        float a2 = in[i];
        out[i] = w[d * 3 + 0] * a0 + w[d * 3 + 1] * a1 + w[d * 3 + 2] * a2 + cb[d];
    }
}

// ---------------------------------------------------------------------------
// Decay mixing as a linear recurrence: s[t] = decay*s[t-1] + step*delta[t]
// One thread per (b,d): 2048 scans of length 2048, coalesced over d.
// ---------------------------------------------------------------------------
__global__ void scan_decay(const float* __restrict__ delta,
                           const float* __restrict__ p_logit,
                           const float* __restrict__ p_step,
                           float* __restrict__ out) {
    const int T = 2048, D = 512;
    const int i = blockIdx.x * blockDim.x + threadIdx.x;
    if (i >= 4 * D) return;
    const int b = i / D, d = i % D;
    const float decay = 1.0f / (1.0f + __expf(-p_logit[0]));
    const float step  = p_step[0];
    const float* dp = delta + (size_t)b * T * D + d;
    float* op = out + (size_t)b * T * D + d;
    float s = 0.0f;
    for (int t = 0; t < T; ++t) {
        s = decay * s + step * dp[(size_t)t * D];
        op[(size_t)t * D] = s;
    }
}

// ---------------------------------------------------------------------------
// Host orchestration
// ---------------------------------------------------------------------------
extern "C" void kernel_launch(void* const* d_in, const int* in_sizes, int n_in,
                              void* d_out, int out_size, void* d_ws, size_t ws_size,
                              hipStream_t stream) {
    (void)in_sizes; (void)n_in; (void)out_size; (void)ws_size;
    const int B = 4, T = 2048, D = 512;
    const int BT = B * T;                 // 8192 rows

    const float* x        = (const float*)d_in[0];
    const float* Wqkv     = (const float*)d_in[1];
    const float* bqkv     = (const float*)d_in[2];
    const float* Wo       = (const float*)d_in[3];
    const float* bo       = (const float*)d_in[4];
    const float* ln1_g    = (const float*)d_in[5];
    const float* ln1_b    = (const float*)d_in[6];
    const float* Wsp      = (const float*)d_in[7];
    const float* bsp      = (const float*)d_in[8];
    const float* conv_w   = (const float*)d_in[9];
    const float* conv_b   = (const float*)d_in[10];
    const float* Wt1      = (const float*)d_in[11];
    const float* bt1      = (const float*)d_in[12];
    const float* Wt2      = (const float*)d_in[13];
    const float* bt2      = (const float*)d_in[14];
    const float* seqln_g  = (const float*)d_in[15];
    const float* seqln_b  = (const float*)d_in[16];
    const float* decay_lg = (const float*)d_in[17];
    const float* step_sz  = (const float*)d_in[18];
    const float* ln2_g    = (const float*)d_in[19];
    const float* ln2_b    = (const float*)d_in[20];
    const float* Wf1      = (const float*)d_in[21];
    const float* bf1      = (const float*)d_in[22];
    const float* Wf2      = (const float*)d_in[23];
    const float* bf2      = (const float*)d_in[24];
    const float* ln3_g    = (const float*)d_in[25];
    const float* ln3_b    = (const float*)d_in[26];
    float* out = (float*)d_out;

    // Workspace layout (byte offsets; lifetime-aliased, ~194 MB total):
    char* ws = (char*)d_ws;
    __bf16* ABF = (__bf16*)(ws + 0);                 // [0,32MB) activation bf16 staging
    __bf16* WT  = (__bf16*)(ws + (32ull << 20));     // [32,42MB) packed weights
    __bf16* WqkvT = WT;
    __bf16* WoT   = WqkvT + (size_t)1536 * 512;
    __bf16* WspT  = WoT   + (size_t)512 * 512;
    __bf16* Wt1T  = WspT  + (size_t)512 * 512;
    __bf16* Wt2T  = Wt1T  + (size_t)1024 * 512;
    __bf16* Wf1T  = Wt2T  + (size_t)512 * 1024;
    __bf16* Wf2T  = Wf1T  + (size_t)2048 * 512;
    float* qkv = (float*)(ws + (42ull << 20));       // [42,90MB)  (dead after pack_qkv)
    float* h1  = (float*)(ws + (42ull << 20));       // [42,74MB)  (dead after convert)
    float* ffh = (float*)(ws + (42ull << 20));       // [42,106MB)
    __bf16* Qb = (__bf16*)(ws + (106ull << 20));     // 8MB each
    __bf16* Kb = Qb + (size_t)B * 8 * T * 64;
    __bf16* Vt = Kb + (size_t)B * 8 * T * 64;        // ends at 130MB
    float* P0 = (float*)(ws + (130ull << 20));       // ao, later x2
    float* P1 = (float*)(ws + (146ull << 20));       // y, then x1 (in-place LN)
    float* P2 = (float*)(ws + (162ull << 20));       // seq_in -> delta -> ff
    float* P3 = (float*)(ws + (178ull << 20));       // seq_conv -> seq_out

    const size_t nBTD  = (size_t)BT * D;             // 4.19M
    const int EW = 256;
    auto blocks = [](size_t n, int bs) { return (unsigned)((n + bs - 1) / bs); };

    // ---- pack weights to [N,K] bf16 ----
    pack_wt<<<blocks((size_t)512 * 1536, EW), EW, 0, stream>>>(Wqkv, WqkvT, 512, 1536);
    pack_wt<<<blocks((size_t)512 * 512,  EW), EW, 0, stream>>>(Wo,   WoT,   512, 512);
    pack_wt<<<blocks((size_t)512 * 512,  EW), EW, 0, stream>>>(Wsp,  WspT,  512, 512);
    pack_wt<<<blocks((size_t)512 * 1024, EW), EW, 0, stream>>>(Wt1,  Wt1T,  512, 1024);
    pack_wt<<<blocks((size_t)1024 * 512, EW), EW, 0, stream>>>(Wt2,  Wt2T,  1024, 512);
    pack_wt<<<blocks((size_t)512 * 2048, EW), EW, 0, stream>>>(Wf1,  Wf1T,  512, 2048);
    pack_wt<<<blocks((size_t)2048 * 512, EW), EW, 0, stream>>>(Wf2,  Wf2T,  2048, 512);

    // ---- attention ----
    f32_to_bf16<<<blocks(nBTD, EW), EW, 0, stream>>>(x, ABF, nBTD);
    gemm_bf16<<<dim3(1536 / 64, BT / 128), 256, 0, stream>>>(ABF, WqkvT, bqkv, qkv, BT, 1536, 512, 0);
    pack_qkv<<<blocks(nBTD, EW), EW, 0, stream>>>(qkv, Qb, Kb, Vt);
    flash_attn<<<dim3(T / 128, B * 8), 256, 0, stream>>>(Qb, Kb, Vt, P0);   // ao -> P0
    f32_to_bf16<<<blocks(nBTD, EW), EW, 0, stream>>>(P0, ABF, nBTD);
    gemm_bf16<<<dim3(512 / 64, BT / 128), 256, 0, stream>>>(ABF, WoT, bo, P1, BT, 512, 512, 0);
    ln_fused<<<BT, 128, 0, stream>>>(P1, x, ln1_g, ln1_b, P1);              // x1 -> P1

    // ---- decay-mixing branch ----
    f32_to_bf16<<<blocks(nBTD, EW), EW, 0, stream>>>(P1, ABF, nBTD);
    gemm_bf16<<<dim3(512 / 64, BT / 128), 256, 0, stream>>>(ABF, WspT, bsp, P2, BT, 512, 512, 0);
    dwconv3<<<blocks(nBTD, EW), EW, 0, stream>>>(P2, conv_w, conv_b, P3);   // seq_conv -> P3
    f32_to_bf16<<<blocks(nBTD, EW), EW, 0, stream>>>(P3, ABF, nBTD);
    gemm_bf16<<<dim3(1024 / 64, BT / 128), 256, 0, stream>>>(ABF, Wt1T, bt1, h1, BT, 1024, 512, 1);
    f32_to_bf16<<<blocks((size_t)BT * 1024, EW), EW, 0, stream>>>(h1, ABF, (size_t)BT * 1024);
    gemm_bf16<<<dim3(512 / 64, BT / 128), 256, 0, stream>>>(ABF, Wt2T, bt2, P2, BT, 512, 1024, 0); // delta
    scan_decay<<<blocks((size_t)B * D, EW), EW, 0, stream>>>(P2, decay_lg, step_sz, P3);           // seq_out
    ln_fused<<<BT, 128, 0, stream>>>(P3, nullptr, seqln_g, seqln_b, P3);    // seqln
    ln_fused<<<BT, 128, 0, stream>>>(P1, P3, ln2_g, ln2_b, P0);             // x2 -> P0

    // ---- FFN ----
    f32_to_bf16<<<blocks(nBTD, EW), EW, 0, stream>>>(P0, ABF, nBTD);
    gemm_bf16<<<dim3(2048 / 64, BT / 128), 256, 0, stream>>>(ABF, Wf1T, bf1, ffh, BT, 2048, 512, 1);
    f32_to_bf16<<<blocks((size_t)BT * 2048, EW), EW, 0, stream>>>(ffh, ABF, (size_t)BT * 2048);
    gemm_bf16<<<dim3(512 / 64, BT / 128), 256, 0, stream>>>(ABF, Wf2T, bf2, P2, BT, 512, 2048, 0);
    ln_fused<<<BT, 128, 0, stream>>>(P0, P2, ln3_g, ln3_b, out);            // final LN
}
